// MultiBoxLoss_89343909691796
// MI455X (gfx1250) — compile-verified, matrix operand order checked by reading, not run on hardware
//
#include <hip/hip_runtime.h>
#include <hip/hip_bf16.h>
#include <stdint.h>

// ---------------------------------------------------------------------------
// MultiBoxLoss for MI455X (gfx1250, wave32).
//
// Roofline: ~119MB mandatory reads -> ~5us at 23.3TB/s; IoU compute ~1.6GFLOP
// of fp32 VALU overlaps with memory. The reference's per-row sort of 131072
// BCE values is replaced by a 2048-bin histogram top-K selection (exact bins
// above threshold + fractional boundary bin), one streaming pass.
//
// CDNA5 paths: TDM tensor_load_to_lds for LDS staging (s_wait_tensorcnt),
// exact IU8 WMMA ones-reduction for positive counting, wave32 shuffles,
// global_prefetch.
// ---------------------------------------------------------------------------

#define BATCH 32
#define NPRI  131072
#define NOBJ  32
#define HBINS 2048
#define HSCALE 64.0f          // bin width = 1/64, range [0,32), clamp above
#define PRI_PER_BLOCK 8192    // 16 tiles per image, 256 threads, 8 waves

typedef __attribute__((ext_vector_type(4))) unsigned int u32x4;
typedef __attribute__((ext_vector_type(8))) int i32x8;
typedef __attribute__((ext_vector_type(4))) int i32x4;

#if defined(__gfx1250__) && __has_builtin(__builtin_amdgcn_tensor_load_to_lds)
#define HAS_TDM 1
#else
#define HAS_TDM 0
#endif

#if defined(__gfx1250__) && __has_builtin(__builtin_amdgcn_wmma_i32_16x16x64_iu8)
#define HAS_WMMA 1
#else
#define HAS_WMMA 0
#endif

// Workspace layout (bytes); requires ws_size >= ~533KB.
#define WS_ACC   0        // float accums[3]: l1_total, posbce_total, hard_total
#define WS_NPOS  256      // int npos[B]
#define WS_KEYS  512      // u64 keys[B*O]
#define WS_HCNT  8704     // u32 hcnt[B*HBINS]
#define WS_HSUM  270848   // f32 hsum[B*HBINS]
#define WS_TOTAL 532992
#define WS_WORDS (WS_TOTAL / 4)

#if HAS_TDM
// 1-D TDM load of `nfloats` contiguous f32 from global to LDS.
// D# per CDNA5 ISA: group0 = {count|flags, lds_addr,
// global_addr[31:0], global_addr[56:32]|type=2}; group1 carries data_size,
// tensor/tile dims and strides. Generic LDS pointer low 32 bits == LDS byte
// offset (ISA §10.2 aperture truncation).
__device__ __forceinline__ void tdm_load_1d(const void* gptr, void* lptr,
                                            unsigned nfloats) {
  unsigned long long ga = (unsigned long long)(uintptr_t)gptr;
  unsigned lds = (unsigned)(uintptr_t)lptr;
  u32x4 g0;
  g0[0] = 1u;                                         // count=1 (valid user D#)
  g0[1] = lds;                                        // lds_addr
  g0[2] = (unsigned)(ga & 0xFFFFFFFFull);             // global_addr[31:0]
  g0[3] = (unsigned)((ga >> 32) & 0x01FFFFFFull) | (2u << 30); // [56:32]+type=2
  i32x8 g1;
  g1[0] = 0x00020000;                                 // data_size=4B, mask=0
  g1[1] = (int)((nfloats & 0xFFFFu) << 16);           // tensor_dim0[15:0]
  g1[2] = (int)((nfloats >> 16) | (1u << 16));        // dim0[31:16], tensor_dim1=1
  g1[3] = (int)((nfloats & 0xFFFFu) << 16);           // tile_dim0
  g1[4] = 1;                                          // tile_dim1=1
  g1[5] = (int)nfloats;                               // tensor_dim0_stride[31:0]
  g1[6] = (int)((nfloats & 0xFFFFu) << 16);           // dim1_stride[15:0]
  g1[7] = 0;
  i32x4 z4 = {0, 0, 0, 0};
#if defined(__clang_major__) && (__clang_major__ >= 23)
  i32x8 z8 = {0, 0, 0, 0, 0, 0, 0, 0};
  __builtin_amdgcn_tensor_load_to_lds(g0, g1, z4, z4, z8, 0);
#else
  __builtin_amdgcn_tensor_load_to_lds(g0, g1, z4, z4, 0);
#endif
}
#endif

__device__ __forceinline__ float softplus_f(float x) {
  return fmaxf(x, 0.0f) + log1pf(expf(-fabsf(x)));
}
__device__ __forceinline__ float wred_f(float v) {
  for (int m = 16; m; m >>= 1) v += __shfl_xor(v, m, 32);
  return v;
}
__device__ __forceinline__ int wred_i(int v) {
  for (int m = 16; m; m >>= 1) v += __shfl_xor(v, m, 32);
  return v;
}

// --------------------------- K0: zero workspace ----------------------------
__global__ void k0_zero(unsigned int* w, int n) {
  int i = blockIdx.x * 256 + threadIdx.x;
  if (i < n) w[i] = 0u;
}

// ------------- K1: per-object best prior (argmax over P) -------------------
// key = (float_bits(iou) << 32) | (0x7FFFFFFF - p): monotone in iou (iou>=0),
// ties resolved toward the LOWER prior index (matches jnp.argmax).
__global__ void __launch_bounds__(256)
k1_best_prior(const float* __restrict__ boxes, const float* __restrict__ priors,
              unsigned long long* __restrict__ keys) {
  const int b = blockIdx.y, tile = blockIdx.x;
  const int tid = threadIdx.x, wave = tid >> 5, lane = tid & 31;
  __shared__ float s_pr[256 * 4];

  // This wave owns objects 4*wave .. 4*wave+3.
  float ox0[4], oy0[4], ox1[4], oy1[4], oarea[4];
#pragma unroll
  for (int j = 0; j < 4; ++j) {
    const float* bp = boxes + ((size_t)b * NOBJ + wave * 4 + j) * 4;
    ox0[j] = bp[0]; oy0[j] = bp[1]; ox1[j] = bp[2]; oy1[j] = bp[3];
    oarea[j] = (ox1[j] - ox0[j]) * (oy1[j] - oy0[j]);
  }
  float bestv[4] = {-1.f, -1.f, -1.f, -1.f};
  int bestp[4] = {0, 0, 0, 0};

  const int pbase = tile * PRI_PER_BLOCK;
  for (int ch = 0; ch < PRI_PER_BLOCK / 256; ++ch) {
    const int cbase = pbase + ch * 256;
    __syncthreads();
#if HAS_TDM
    if (wave == 0) {  // wave-uniform branch; TDM ignores EXEC, one DMA issued
      tdm_load_1d(priors + (size_t)cbase * 4, s_pr, 256 * 4);
      __builtin_amdgcn_s_wait_tensorcnt(0);
    }
#else
    ((float4*)s_pr)[tid] = ((const float4*)priors)[cbase + tid];
#endif
    __syncthreads();
#pragma unroll
    for (int i = 0; i < 8; ++i) {
      const int idx = i * 32 + lane;
      const float pcx = s_pr[idx * 4], pcy = s_pr[idx * 4 + 1];
      const float pw = s_pr[idx * 4 + 2], ph = s_pr[idx * 4 + 3];
      const float px0 = pcx - 0.5f * pw, py0 = pcy - 0.5f * ph;
      const float px1 = pcx + 0.5f * pw, py1 = pcy + 0.5f * ph;
      const float parea = pw * ph;
      const int p = cbase + idx;
#pragma unroll
      for (int j = 0; j < 4; ++j) {
        float ix = fmaxf(0.f, fminf(ox1[j], px1) - fmaxf(ox0[j], px0));
        float iy = fmaxf(0.f, fminf(oy1[j], py1) - fmaxf(oy0[j], py0));
        float inter = ix * iy;
        float iou = inter / (oarea[j] + parea - inter);
        if (iou > bestv[j]) { bestv[j] = iou; bestp[j] = p; }
      }
    }
  }
#pragma unroll
  for (int j = 0; j < 4; ++j) {
    float v = bestv[j] < 0.f ? 0.f : bestv[j];
    unsigned long long key =
        ((unsigned long long)__float_as_uint(v) << 32) |
        (unsigned)(0x7FFFFFFF - bestp[j]);
    for (int m = 16; m; m >>= 1) {
      unsigned long long o = __shfl_xor(key, m, 32);
      if (o > key) key = o;
    }
    if (lane == 0)
      atomicMax(&keys[b * NOBJ + wave * 4 + j], key);
  }
}

// --------- K3: fused match + BCE + L1 + histogram + WMMA pos-count ---------
__global__ void __launch_bounds__(256)
k3_loss(const float* __restrict__ plocs, const float* __restrict__ pscores,
        const float* __restrict__ boxes, const int* __restrict__ labels,
        const float* __restrict__ priors,
        const unsigned long long* __restrict__ keys, int* __restrict__ npos_g,
        unsigned int* __restrict__ hcnt_g, float* __restrict__ hsum_g,
        float* __restrict__ accums) {
  const int b = blockIdx.y, tile = blockIdx.x;
  const int tid = threadIdx.x, wave = tid >> 5, lane = tid & 31;
  __shared__ float s_bx[NOBJ * 4];
  __shared__ float s_area[NOBJ];
  __shared__ int s_lb[NOBJ];
  __shared__ int s_win[NOBJ];
  __shared__ unsigned int s_hc[HBINS];
  __shared__ float s_hs[HBINS];

  for (int i = tid; i < HBINS; i += 256) { s_hc[i] = 0u; s_hs[i] = 0.f; }
#if HAS_TDM
  if (wave == 0) {
    tdm_load_1d(boxes + (size_t)b * NOBJ * 4, s_bx, NOBJ * 4);
    __builtin_amdgcn_s_wait_tensorcnt(0);
  }
#else
  if (tid < NOBJ * 4) s_bx[tid] = boxes[(size_t)b * NOBJ * 4 + tid];
#endif
  if (tid < NOBJ) {
    s_lb[tid] = labels[b * NOBJ + tid];
    unsigned long long k = keys[b * NOBJ + tid];
    s_win[tid] = 0x7FFFFFFF - (int)(unsigned)(k & 0xFFFFFFFFull);
  }
  __syncthreads();
  if (tid < NOBJ)
    s_area[tid] = (s_bx[tid * 4 + 2] - s_bx[tid * 4]) *
                  (s_bx[tid * 4 + 3] - s_bx[tid * 4 + 1]);
  __syncthreads();

  const int base = tile * PRI_PER_BLOCK + wave * 1024;
  unsigned int aw[8] = {0, 0, 0, 0, 0, 0, 0, 0};  // WMMA A: 32 flag bytes/lane
  float l1 = 0.f, posbce = 0.f;

#pragma unroll 4
  for (int i = 0; i < 32; ++i) {
    const int p = base + i * 32 + lane;
    const float4 pr = ((const float4*)priors)[p];
    const float px0 = pr.x - 0.5f * pr.z, py0 = pr.y - 0.5f * pr.w;
    const float px1 = pr.x + 0.5f * pr.z, py1 = pr.y + 0.5f * pr.w;
    const float parea = pr.z * pr.w;

    float maxov = -1.f; int argo = 0;
#pragma unroll
    for (int o = 0; o < NOBJ; ++o) {
      float ix = fmaxf(0.f, fminf(s_bx[o * 4 + 2], px1) - fmaxf(s_bx[o * 4], px0));
      float iy = fmaxf(0.f, fminf(s_bx[o * 4 + 3], py1) - fmaxf(s_bx[o * 4 + 1], py0));
      float inter = ix * iy;
      float iou = inter / (s_area[o] + parea - inter);
      if (iou > maxov) { maxov = iou; argo = o; }
    }
    // forced best-prior-per-object overwrite (last duplicate wins)
#pragma unroll
    for (int o = 0; o < NOBJ; ++o)
      if (s_win[o] == p) { argo = o; maxov = 1.0f; }

    const int lbl = (maxov < 0.5f) ? 0 : s_lb[argo];

    const float* sc = pscores + ((size_t)b * NPRI + p) * 3;
    __builtin_prefetch(sc + 32 * 3, 0, 0);  // next i-iteration -> global_prefetch
    const float s0 = sc[0], s1 = sc[1], s2 = sc[2];
    // targets: 0->(1,0,0) 1->(0,1,0) 2->(0,0,1) 3->(0,1,1)
    const float t0 = (lbl == 0) ? 1.f : 0.f;
    const float t1 = (lbl == 1 || lbl == 3) ? 1.f : 0.f;
    const float t2 = (lbl == 2 || lbl == 3) ? 1.f : 0.f;
    const float bce = softplus_f(s0) - s0 * t0 + softplus_f(s1) - s1 * t1 +
                      softplus_f(s2) - s2 * t2;

    if (lbl > 0) {
      aw[i >> 2] |= 1u << ((i & 3) * 8);
      posbce += bce;
      const float4 pl = ((const float4*)plocs)[(size_t)b * NPRI + p];
      const float bx0 = s_bx[argo * 4], by0 = s_bx[argo * 4 + 1];
      const float bx1 = s_bx[argo * 4 + 2], by1 = s_bx[argo * 4 + 3];
      const float bw = bx1 - bx0, bh = by1 - by0;
      const float g0 = ((bx0 + bx1) * 0.5f - pr.x) * 10.f / pr.z;
      const float g1 = ((by0 + by1) * 0.5f - pr.y) * 10.f / pr.w;
      const float g2 = logf(bw / pr.z) * 5.f;
      const float g3 = logf(bh / pr.w) * 5.f;
      l1 += fabsf(pl.x - g0) + fabsf(pl.y - g1) + fabsf(pl.z - g2) +
            fabsf(pl.w - g3);
    } else {
      int bin = (int)(bce * HSCALE);
      bin = bin < 0 ? 0 : (bin > HBINS - 1 ? HBINS - 1 : bin);
      atomicAdd(&s_hc[bin], 1u);
      atomicAdd(&s_hs[bin], bce);
    }
  }

  // Positive count via exact IU8 WMMA ones-reduction: A = 1024 flag bytes of
  // this wave, B = ones => sum over all D entries = 16 * n_pos(wave).
  int wave_cnt;
#if HAS_WMMA
  {
    union { unsigned int w[8]; i32x8 v; } ua;
#pragma unroll
    for (int j = 0; j < 8; ++j) ua.w[j] = aw[j];
    i32x8 ones;
#pragma unroll
    for (int j = 0; j < 8; ++j) ones[j] = 0x01010101;
    i32x8 c0 = {0, 0, 0, 0, 0, 0, 0, 0};
    i32x8 d = __builtin_amdgcn_wmma_i32_16x16x64_iu8(false, ua.v, false, ones,
                                                     c0, false, false);
    int local = d[0] + d[1] + d[2] + d[3] + d[4] + d[5] + d[6] + d[7];
    wave_cnt = wred_i(local) >> 4;
  }
#else
  {
    int local = 0;
#pragma unroll
    for (int j = 0; j < 8; ++j) {
      unsigned w = aw[j];
      local += (w & 0xFF) + ((w >> 8) & 0xFF) + ((w >> 16) & 0xFF) + (w >> 24);
    }
    wave_cnt = wred_i(local);
  }
#endif
  const float l1w = wred_f(l1);
  const float pbw = wred_f(posbce);
  if (lane == 0) {
    if (wave_cnt) atomicAdd(&npos_g[b], wave_cnt);
    if (l1w != 0.f) atomicAdd(&accums[0], l1w);
    if (pbw != 0.f) atomicAdd(&accums[1], pbw);
  }

  __syncthreads();
  for (int i = tid; i < HBINS; i += 256) {
    if (s_hc[i]) {
      atomicAdd(&hcnt_g[b * HBINS + i], s_hc[i]);
      atomicAdd(&hsum_g[b * HBINS + i], s_hs[i]);
    }
  }
}

// ------------- K4: per-row top-K sum from histogram ------------------------
__global__ void k4_mine(const int* __restrict__ npos_g,
                        const unsigned int* __restrict__ hcnt,
                        const float* __restrict__ hsum,
                        float* __restrict__ accums) {
  const int b = blockIdx.x;
  if (threadIdx.x != 0) return;
  const long long K = 3LL * (long long)npos_g[b];
  long long cum = 0;
  float sum = 0.f;
  for (int bin = HBINS - 1; bin >= 0 && cum < K; --bin) {
    const unsigned c = hcnt[b * HBINS + bin];
    if (!c) continue;
    const float s = hsum[b * HBINS + bin];
    if (cum + (long long)c <= K) { sum += s; cum += c; }
    else { sum += s * (float)(K - cum) / (float)c; cum = K; }
  }
  if (sum != 0.f) atomicAdd(&accums[2], sum);
}

// ------------------------------ K5: finalize -------------------------------
__global__ void k5_final(const int* __restrict__ npos_g,
                         const float* __restrict__ accums,
                         float* __restrict__ out) {
  if (threadIdx.x == 0 && blockIdx.x == 0) {
    long long tot = 0;
    for (int b = 0; b < BATCH; ++b) tot += npos_g[b];
    const float l1 = accums[0], pb = accums[1], hs = accums[2];
    const float conf = (hs + pb) / (1e-10f + (float)tot);
    long long d4 = tot * 4; if (d4 < 1) d4 = 1;
    const float loc = (tot > 0) ? l1 / (float)d4 : 0.f;
    out[0] = conf + loc;
    out[1] = conf;
    out[2] = loc;
    for (int b = 0; b < BATCH; ++b) out[3 + b] = (float)npos_g[b];
  }
}

extern "C" void kernel_launch(void* const* d_in, const int* in_sizes, int n_in,
                              void* d_out, int out_size, void* d_ws,
                              size_t ws_size, hipStream_t stream) {
  const float* plocs = (const float*)d_in[0];    // [B,P,4]
  const float* pscores = (const float*)d_in[1];  // [B,P,3]
  const float* boxes = (const float*)d_in[2];    // [B,O,4] xy
  const int* labels = (const int*)d_in[3];       // [B,O]
  const float* priors = (const float*)d_in[4];   // [P,4] cxcy
  float* out = (float*)d_out;

  char* ws = (char*)d_ws;  // requires ws_size >= WS_TOTAL (~533KB)
  float* accums = (float*)(ws + WS_ACC);
  int* npos = (int*)(ws + WS_NPOS);
  unsigned long long* keys = (unsigned long long*)(ws + WS_KEYS);
  unsigned int* hcnt = (unsigned int*)(ws + WS_HCNT);
  float* hsum = (float*)(ws + WS_HSUM);

  k0_zero<<<(WS_WORDS + 255) / 256, 256, 0, stream>>>((unsigned int*)ws,
                                                      WS_WORDS);
  dim3 grid(NPRI / PRI_PER_BLOCK, BATCH), blk(256);
  k1_best_prior<<<grid, blk, 0, stream>>>(boxes, priors, keys);
  k3_loss<<<grid, blk, 0, stream>>>(plocs, pscores, boxes, labels, priors,
                                    keys, npos, hcnt, hsum, accums);
  k4_mine<<<BATCH, 32, 0, stream>>>(npos, hcnt, hsum, accums);
  k5_final<<<1, 32, 0, stream>>>(npos, accums, out);
}
